// DirectedAcyclicDecoder_47210280517870
// MI455X (gfx1250) — matrix operand order repaired and checked by reading, not compiled, not run on hardware
//
#include <hip/hip_runtime.h>

typedef __attribute__((ext_vector_type(16))) __bf16 v16bf;
typedef __attribute__((ext_vector_type(8)))  __bf16 v8bf;
typedef __attribute__((ext_vector_type(8)))  float  v8f;

#define B_    8
#define T_    1024
#define D_    512
#define H_    8
#define CH_   64
#define TWOD_ 1024
#define NINF (-__builtin_inff())

// ---------------------------------------------------------------------------
// Kernel 1: fused q/k projection.  proj[m][0:512] = q row, proj[m][512:1024] = k row
// q[m,n] = sum_k f[m,k] * Wq[n,k] + bq[n]   (f = [features | pos_embed])
// bf16 WMMA 16x16x32, f32 accumulate, bf16 store.
// ---------------------------------------------------------------------------
__global__ __launch_bounds__(256) void proj_kernel(
    const float* __restrict__ feat, const float* __restrict__ pos,
    const float* __restrict__ Wq, const float* __restrict__ bq,
    const float* __restrict__ Wk, const float* __restrict__ bk,
    __bf16* __restrict__ proj)
{
  __shared__ __align__(64) __bf16 At[128][32];   // 8 KB
  __shared__ __align__(64) __bf16 Bt[64][32];    // 4 KB

  const int tid  = threadIdx.x;
  const int lane = tid & 31;
  const int w    = tid >> 5;            // wave id 0..7 -> 16-row strip
  const int M0   = blockIdx.x * 128;    // rows of [B*T]
  const int N0   = blockIdx.y * 64;     // cols of [q(512) | k(512)]

  v8f acc[4];
  #pragma unroll
  for (int nt = 0; nt < 4; ++nt) acc[nt] = v8f{};

  for (int k0 = 0; k0 < TWOD_; k0 += 32) {
    // stage A tile (f32 -> bf16): 128 x 32
    {
      int r  = tid >> 1;
      int c0 = (tid & 1) * 16;
      int mg = M0 + r;
      int kk = k0 + c0;                 // 16-aligned; never straddles 512
      const float* src = (kk < D_) ? (feat + (size_t)mg * D_ + kk)
                                   : (pos  + (size_t)mg * D_ + (kk - D_));
      #pragma unroll
      for (int e = 0; e < 16; ++e) At[r][c0 + e] = (__bf16)src[e];
    }
    // stage B tile (weights, row-major [n][k]): 64 x 32
    {
      int nn   = tid >> 2;
      int c0   = (tid & 3) * 8;
      int ncol = N0 + nn;
      const float* wrow = (ncol < D_) ? (Wq + (size_t)ncol * TWOD_)
                                      : (Wk + (size_t)(ncol - D_) * TWOD_);
      #pragma unroll
      for (int e = 0; e < 8; ++e) Bt[nn][c0 + e] = (__bf16)wrow[k0 + c0 + e];
    }
    __syncthreads();

    // A fragment: lane l holds row m=l&15; K split: lanes>=16 get K+16..K+31
    const int m = (w << 4) + (lane & 15);
    const int p = (lane >> 4) * 16;
    v16bf a;
    {
      v8bf lo = *(const v8bf*)&At[m][p];
      v8bf hi = *(const v8bf*)&At[m][p + 8];
      #pragma unroll
      for (int e = 0; e < 8; ++e) { a[e] = lo[e]; a[e + 8] = hi[e]; }
    }
    const int kb = (lane >> 4) * 16;
    // preload all 4 B fragments, then issue the 4 WMMAs back-to-back so the
    // ds_loads batch up and the XDL pipe stays busy
    v16bf bfr[4];
    #pragma unroll
    for (int nt = 0; nt < 4; ++nt) {
      int n = nt * 16 + (lane & 15);
      bfr[nt] = *(const v16bf*)&Bt[n][kb];
    }
    #pragma unroll
    for (int nt = 0; nt < 4; ++nt) {
      acc[nt] = __builtin_amdgcn_wmma_f32_16x16x32_bf16(
          false, a, false, bfr[nt], (short)0, acc[nt], false, false);
    }
    __syncthreads();
  }

  // store: C layout row = v + 8*(lane>>4), col = lane&15 within 16x16 tile
  #pragma unroll
  for (int nt = 0; nt < 4; ++nt) {
    int ncol   = N0 + nt * 16 + (lane & 15);
    float bias = (ncol < D_) ? bq[ncol] : bk[ncol - D_];
    #pragma unroll
    for (int v = 0; v < 8; ++v) {
      int row = M0 + (w << 4) + v + ((lane >> 4) << 3);
      proj[(size_t)row * TWOD_ + ncol] = (__bf16)(acc[nt][v] + bias);
    }
  }
}

// ---------------------------------------------------------------------------
// Kernel 2: gate logits + log_softmax over H=8.  One wave per row m.
// ---------------------------------------------------------------------------
__global__ __launch_bounds__(256) void gates_kernel(
    const float* __restrict__ feat, const float* __restrict__ pos,
    const float* __restrict__ Wg, const float* __restrict__ bg,
    float* __restrict__ glog)
{
  const int gw   = (blockIdx.x * blockDim.x + threadIdx.x) >> 5;  // row m
  const int lane = threadIdx.x & 31;
  if (gw >= B_ * T_) return;
  const float* frow = feat + (size_t)gw * D_;
  const float* prow = pos  + (size_t)gw * D_;
  float acc[H_];
  #pragma unroll
  for (int h = 0; h < H_; ++h) acc[h] = 0.f;
  for (int kk = lane; kk < TWOD_; kk += 32) {
    float fv = (kk < D_) ? frow[kk] : prow[kk - D_];
    #pragma unroll
    for (int h = 0; h < H_; ++h) acc[h] += fv * Wg[h * TWOD_ + kk];
  }
  #pragma unroll
  for (int h = 0; h < H_; ++h)
    #pragma unroll
    for (int off = 16; off >= 1; off >>= 1)
      acc[h] += __shfl_xor(acc[h], off, 32);
  if (lane == 0) {
    float mx = NINF;
    #pragma unroll
    for (int h = 0; h < H_; ++h) { acc[h] += bg[h]; mx = fmaxf(mx, acc[h]); }
    float s = 0.f;
    #pragma unroll
    for (int h = 0; h < H_; ++h) s += __expf(acc[h] - mx);
    float lse = mx + __logf(s);
    #pragma unroll
    for (int h = 0; h < H_; ++h) glog[(size_t)gw * H_ + h] = acc[h] - lse;
  }
}

// ---------------------------------------------------------------------------
// Kernel 3: n_valid per batch (tokens != 0)
// ---------------------------------------------------------------------------
__global__ __launch_bounds__(256) void nvalid_kernel(const int* __restrict__ tokens,
                                                     int* __restrict__ nv)
{
  __shared__ int sbuf[256];
  int b = blockIdx.x;
  int cnt = 0;
  for (int t = threadIdx.x; t < T_; t += 256) cnt += (tokens[b * T_ + t] != 0);
  sbuf[threadIdx.x] = cnt;
  __syncthreads();
  for (int s = 128; s > 0; s >>= 1) {
    if (threadIdx.x < s) sbuf[threadIdx.x] += sbuf[threadIdx.x + s];
    __syncthreads();
  }
  if (threadIdx.x == 0) nv[b] = sbuf[0];
}

// ---------------------------------------------------------------------------
// Kernel 4: fill output with -inf
// ---------------------------------------------------------------------------
__global__ __launch_bounds__(256) void fill_kernel(float4* __restrict__ out, int n4)
{
  int i = blockIdx.x * 256 + threadIdx.x;
  if (i < n4) out[i] = float4{NINF, NINF, NINF, NINF};
}

// ---------------------------------------------------------------------------
// Kernel 5: fused scores + band log_softmax(j) + head-gated logsumexp.
// Block = (batch b, 16-row i-tile).  Wave w = head h.  Two-pass online
// softmax over j; scores recomputed in pass 2 (K stays in L2).
// ---------------------------------------------------------------------------
__global__ __launch_bounds__(256) void scores_kernel(
    const __bf16* __restrict__ proj, const float* __restrict__ glog,
    const int* __restrict__ nvalid, float* __restrict__ out)
{
  __shared__ float lse_s[H_][16];
  __shared__ float lg_s[16][H_];
  __shared__ float tbuf[H_][16][16];   // 8 KB

  const int tid  = threadIdx.x;
  const int lane = tid & 31;
  const int h    = tid >> 5;                  // head = wave id
  const int bIdx = blockIdx.x >> 6;           // 64 i-tiles per batch
  const int i0   = (blockIdx.x & 63) << 4;

  if (tid < 128) {
    int rr = tid >> 3, hh = tid & 7;
    lg_s[rr][hh] = glog[((size_t)(bIdx * T_ + i0 + rr)) * H_ + hh];
  }
  const int nv = nvalid[bIdx];
  __syncthreads();

  // q fragments for rows i0..i0+15, head h, K=64 -> two 16x16x32 fragments
  const __bf16* qrow = proj + ((size_t)(bIdx * T_ + i0 + (lane & 15))) * TWOD_ + h * CH_;
  const int p = (lane >> 4) * 16;
  v16bf a0, a1;
  {
    v8bf c0 = *(const v8bf*)(qrow + p);
    v8bf c1 = *(const v8bf*)(qrow + p + 8);
    v8bf c2 = *(const v8bf*)(qrow + p + 32);
    v8bf c3 = *(const v8bf*)(qrow + p + 40);
    #pragma unroll
    for (int e = 0; e < 8; ++e) {
      a0[e] = c0[e]; a0[e + 8] = c1[e];
      a1[e] = c2[e]; a1[e + 8] = c3[e];
    }
  }

  const int jt0 = i0 >> 4;                    // tiles fully below band skipped
  const int kb  = (lane >> 4) * 16;
  const __bf16* kbase = proj + ((size_t)(bIdx * T_ + (lane & 15))) * TWOD_
                      + D_ + h * CH_ + kb;

  float M[8], S[8];
  #pragma unroll
  for (int v = 0; v < 8; ++v) { M[v] = NINF; S[v] = 0.f; }

  // ---- pass 1: online max/sum over the band -----------------------------
  for (int jt = jt0; jt < 64; ++jt) {
    const __bf16* krow = kbase + ((size_t)(jt << 4)) * TWOD_;
    if (jt + 1 < 64) {                         // overlap next tile's L2 fetch
      __builtin_prefetch(krow + (size_t)16 * TWOD_, 0, 1);
      __builtin_prefetch(krow + (size_t)16 * TWOD_ + 32, 0, 1);
    }
    v16bf b0 = *(const v16bf*)(krow);
    v16bf b1 = *(const v16bf*)(krow + 32);
    v8f c = v8f{};
    c = __builtin_amdgcn_wmma_f32_16x16x32_bf16(false, a0, false, b0, (short)0, c, false, false);
    c = __builtin_amdgcn_wmma_f32_16x16x32_bf16(false, a1, false, b1, (short)0, c, false, false);
    const int j = (jt << 4) + (lane & 15);
    #pragma unroll
    for (int v = 0; v < 8; ++v) {
      int iRow = i0 + v + ((lane >> 4) << 3);
      float x = (j > iRow && j < nv) ? c[v] * 0.125f : NINF;
      float tm = x;
      #pragma unroll
      for (int off = 8; off >= 1; off >>= 1) tm = fmaxf(tm, __shfl_xor(tm, off, 32));
      float nM = fmaxf(M[v], tm);
      if (nM != NINF) {
        float e = (x == NINF) ? 0.f : __expf(x - nM);
        #pragma unroll
        for (int off = 8; off >= 1; off >>= 1) e += __shfl_xor(e, off, 32);
        S[v] = S[v] * __expf(M[v] - nM) + e;   // exp(-inf)=0 when M was -inf
        M[v] = nM;
      }
    }
  }
  #pragma unroll
  for (int v = 0; v < 8; ++v) {
    float l = (M[v] == NINF) ? NINF : (M[v] + __logf(S[v]));
    if ((lane & 15) == 0) lse_s[h][v + ((lane >> 4) << 3)] = l;
  }
  __syncthreads();

  // ---- pass 2: recompute, gate, logsumexp over heads, write band --------
  for (int jt = jt0; jt < 64; ++jt) {
    const __bf16* krow = kbase + ((size_t)(jt << 4)) * TWOD_;
    if (jt + 1 < 64) {
      __builtin_prefetch(krow + (size_t)16 * TWOD_, 0, 1);
      __builtin_prefetch(krow + (size_t)16 * TWOD_ + 32, 0, 1);
    }
    v16bf b0 = *(const v16bf*)(krow);
    v16bf b1 = *(const v16bf*)(krow + 32);
    v8f c = v8f{};
    c = __builtin_amdgcn_wmma_f32_16x16x32_bf16(false, a0, false, b0, (short)0, c, false, false);
    c = __builtin_amdgcn_wmma_f32_16x16x32_bf16(false, a1, false, b1, (short)0, c, false, false);
    const int j = (jt << 4) + (lane & 15);
    #pragma unroll
    for (int v = 0; v < 8; ++v) {
      int m    = v + ((lane >> 4) << 3);
      int iRow = i0 + m;
      float x = (j > iRow && j < nv) ? c[v] * 0.125f : NINF;
      float L = lse_s[h][m];
      float t = (x == NINF || L == NINF) ? NINF : (x - L + lg_s[m][h]);
      tbuf[h][m][lane & 15] = t;
    }
    __syncthreads();
    {
      int m = tid >> 4, n = tid & 15;
      float vals[H_];
      float mx = NINF;
      #pragma unroll
      for (int hh = 0; hh < H_; ++hh) { vals[hh] = tbuf[hh][m][n]; mx = fmaxf(mx, vals[hh]); }
      float r;
      if (mx == NINF) r = NINF;
      else {
        float s = 0.f;
        #pragma unroll
        for (int hh = 0; hh < H_; ++hh)
          s += (vals[hh] == NINF) ? 0.f : __expf(vals[hh] - mx);
        r = mx + __logf(s);
      }
      out[((size_t)(bIdx * T_ + i0 + m)) * T_ + (jt << 4) + n] = r;
    }
    __syncthreads();
  }
}

// ---------------------------------------------------------------------------
extern "C" void kernel_launch(void* const* d_in, const int* in_sizes, int n_in,
                              void* d_out, int out_size, void* d_ws, size_t ws_size,
                              hipStream_t stream) {
  const float* feat = (const float*)d_in[0];
  const float* pos  = (const float*)d_in[1];
  const int*   tok  = (const int*)d_in[2];
  const float* Wq   = (const float*)d_in[3];
  const float* bq   = (const float*)d_in[4];
  const float* Wk   = (const float*)d_in[5];
  const float* bk   = (const float*)d_in[6];
  const float* Wg   = (const float*)d_in[7];
  const float* bg   = (const float*)d_in[8];
  float* out = (float*)d_out;

  // workspace layout
  __bf16* proj = (__bf16*)d_ws;                                    // 16 MB
  float*  glog = (float*)((char*)d_ws + (size_t)16 * 1024 * 1024); // 256 KB
  int*    nv   = (int*)((char*)d_ws + (size_t)16 * 1024 * 1024 + 262144);

  nvalid_kernel<<<B_, 256, 0, stream>>>(tok, nv);
  gates_kernel<<<(B_ * T_ * 32) / 256, 256, 0, stream>>>(feat, pos, Wg, bg, glog);
  {
    int n4 = (B_ * T_ * T_) / 4;
    fill_kernel<<<(n4 + 255) / 256, 256, 0, stream>>>((float4*)out, n4);
  }
  proj_kernel<<<dim3((B_ * T_) / 128, TWOD_ / 64), 256, 0, stream>>>(
      feat, pos, Wq, bq, Wk, bk, proj);
  scores_kernel<<<B_ * (T_ / 16), 256, 0, stream>>>(proj, glog, nv, out);
}
